// MultiHeadSelfCrossAttention_7834020348638
// MI455X (gfx1250) — compile-verified, matrix operand order checked by reading, not run on hardware
//
#include <hip/hip_runtime.h>

typedef __attribute__((ext_vector_type(16))) _Float16 v16h;
typedef __attribute__((ext_vector_type(8)))  _Float16 v8h;
typedef __attribute__((ext_vector_type(8)))  float    v8f;

#define DEV static __device__ __forceinline__

static constexpr int BDIM = 2;       // batch
static constexpr int SLEN = 2048;    // sequence
static constexpr int MROWS = BDIM * SLEN;  // 4096 GEMM rows

DEV v8f vzero() {
  v8f v;
#pragma unroll
  for (int i = 0; i < 8; ++i) v[i] = 0.f;
  return v;
}

DEV v8f wmma_f16(v16h a, v16h b, v8f c) {
  return __builtin_amdgcn_wmma_f32_16x16x32_f16(
      /*neg_a=*/false, a, /*neg_b=*/false, b,
      /*c_mod=*/(short)0, c, /*reuse_a=*/false, /*reuse_b=*/false);
}

// Load a 16x32 f16 A-style fragment from row-major [., ld] at (row0, k0).
// Lane ln=lane&15 owns row row0+ln; hi=lane>>4 selects K-halves:
//   elems 0..7  -> k0 + hi*8 + 0..7
//   elems 8..15 -> k0 + 16 + hi*8 + 0..7
// B fragments use the symmetric layout (lane owns column n), so passing the
// row-major weight matrix W[N,K] (or K[S,DH], V^T[DH,S]) here gives B = W^T.
DEV v16h load_frag16x32(const _Float16* __restrict__ base, int row0, int ld, int k0) {
  int lane = threadIdx.x & 31;
  int ln = lane & 15, hi = lane >> 4;
  const _Float16* p = base + (size_t)(row0 + ln) * (size_t)ld + (size_t)(k0 + hi * 8);
  v8h lo = *(const v8h*)p;
  v8h hv = *(const v8h*)(p + 16);
  v16h r;
#pragma unroll
  for (int i = 0; i < 8; ++i) { r[i] = lo[i]; r[i + 8] = hv[i]; }
  return r;
}

// ---------------------------------------------------------------- conversion
__global__ void __launch_bounds__(256)
cvt_f32_f16(const float* __restrict__ in, _Float16* __restrict__ out, int n) {
  int i = blockIdx.x * blockDim.x + threadIdx.x;
  if (i < n) out[i] = (_Float16)in[i];
}

// ------------------------------------------------------- GEMM: relu(A*W^T+b)
// A: f16 [M, lda]; W: f16 [N, ldw] row-major (so dot of rows); bias f32 [N].
// Block = 256 thr = 8 waves arranged 4(M)x2(N); wave tile 32x64; block 128x128.
template <int OUT_HALF>
__global__ void __launch_bounds__(256)
gemm_bias_relu(const _Float16* __restrict__ A, int lda,
               const _Float16* __restrict__ W, int ldw,
               const float* __restrict__ bias,
               void* __restrict__ Cv, int ldc, int K) {
  int lane = threadIdx.x & 31;
  int ln = lane & 15, hi = lane >> 4;
  int w = threadIdx.x >> 5;
  int row0 = blockIdx.x * 128 + (w >> 1) * 32;
  int col0 = blockIdx.y * 128 + (w & 1) * 64;

  v8f acc[2][4];
#pragma unroll
  for (int mt = 0; mt < 2; ++mt)
#pragma unroll
    for (int nt = 0; nt < 4; ++nt) acc[mt][nt] = vzero();

  for (int k = 0; k < K; k += 32) {
    v16h af[2], bf[4];
#pragma unroll
    for (int mt = 0; mt < 2; ++mt) af[mt] = load_frag16x32(A, row0 + mt * 16, lda, k);
#pragma unroll
    for (int nt = 0; nt < 4; ++nt) bf[nt] = load_frag16x32(W, col0 + nt * 16, ldw, k);
#pragma unroll
    for (int mt = 0; mt < 2; ++mt)
#pragma unroll
      for (int nt = 0; nt < 4; ++nt) acc[mt][nt] = wmma_f16(af[mt], bf[nt], acc[mt][nt]);
  }

#pragma unroll
  for (int nt = 0; nt < 4; ++nt) {
    int col = col0 + nt * 16 + ln;
    float bv = bias[col];
#pragma unroll
    for (int mt = 0; mt < 2; ++mt) {
#pragma unroll
      for (int r = 0; r < 8; ++r) {
        int row = row0 + mt * 16 + r + 8 * hi;
        float val = fmaxf(acc[mt][nt][r] + bv, 0.f);
        if (OUT_HALF)
          ((_Float16*)Cv)[(size_t)row * ldc + col] = (_Float16)val;
        else
          ((float*)Cv)[(size_t)row * ldc + col] = val;
      }
    }
  }
}

// ------------------------------------------------- V transpose for PV B-frags
// Y slice (cols colOff..colOff+511 of [B*S, 2048]) -> Vt[(b*8+h)*64 + d][s]
__global__ void __launch_bounds__(256)
vtrans(const _Float16* __restrict__ Y, _Float16* __restrict__ Vt, int colOff) {
  int i = blockIdx.x * blockDim.x + threadIdx.x;  // over B*S*512
  if (i >= BDIM * SLEN * 512) return;
  int c = i & 511;
  int s = (i >> 9) & (SLEN - 1);
  int b = i >> 20;
  int h = c >> 6, d = c & 63;
  Vt[(((size_t)(b * 8 + h)) * 64 + d) * SLEN + s] =
      Y[((size_t)(b * SLEN + s)) * 2048 + colOff + c];
}

// ------------------------------------------------------------ flash attention
// Q,K: f16 [B*S, 512] (head h at cols h*64..h*64+63); Vt: f16 [(b*8+h)*64+d][S].
// Each wave: 32 query rows, online softmax over 64 key-blocks of 32.
__global__ void __launch_bounds__(256)
attn_kernel(const _Float16* __restrict__ Qb,
            const _Float16* __restrict__ Kb,
            const _Float16* __restrict__ Vt,
            _Float16* __restrict__ Y, int colOff) {
  __shared__ _Float16 ptile[8 * 32 * 32];  // per-wave 32x32 P staging
  int lane = threadIdx.x & 31;
  int ln = lane & 15, hi = lane >> 4;
  int w = threadIdx.x >> 5;
  int waveTile = blockIdx.x * 8 + w;   // 1024 tiles total
  int stile = waveTile & 63;
  int headLin = waveTile >> 6;         // b*8 + h
  int b = headLin >> 3;
  int h = headLin & 7;
  int s0 = stile * 32;
  int qkCol = h * 64;
  int rowBase = b * SLEN;
  const _Float16* Vh = Vt + (size_t)headLin * 64 * SLEN;
  _Float16* pbase = ptile + w * 1024;

  // Preload Q fragments [mtile][kstep]
  v16h qf[2][2];
#pragma unroll
  for (int mt = 0; mt < 2; ++mt)
#pragma unroll
    for (int ks = 0; ks < 2; ++ks)
      qf[mt][ks] = load_frag16x32(Qb, rowBase + s0 + mt * 16, 512, qkCol + ks * 32);

  v8f acc[2][4];
  float mrow[2][8], lrow[2][8];
#pragma unroll
  for (int mt = 0; mt < 2; ++mt) {
#pragma unroll
    for (int nt = 0; nt < 4; ++nt) acc[mt][nt] = vzero();
#pragma unroll
    for (int r = 0; r < 8; ++r) { mrow[mt][r] = -3.0e38f; lrow[mt][r] = 0.f; }
  }

  for (int kb = 0; kb < SLEN / 32; ++kb) {
    int key0 = kb * 32;
    // ---- S = Q K^T (scaled)
    v8f sacc[2][2];
#pragma unroll
    for (int mt = 0; mt < 2; ++mt)
#pragma unroll
      for (int nt = 0; nt < 2; ++nt) sacc[mt][nt] = vzero();
#pragma unroll
    for (int ks = 0; ks < 2; ++ks) {
      v16h kf0 = load_frag16x32(Kb, rowBase + key0, 512, qkCol + ks * 32);
      v16h kf1 = load_frag16x32(Kb, rowBase + key0 + 16, 512, qkCol + ks * 32);
#pragma unroll
      for (int mt = 0; mt < 2; ++mt) {
        sacc[mt][0] = wmma_f16(qf[mt][ks], kf0, sacc[mt][0]);
        sacc[mt][1] = wmma_f16(qf[mt][ks], kf1, sacc[mt][1]);
      }
    }
    // ---- online softmax (rows live across 16 lanes of one half)
#pragma unroll
    for (int mt = 0; mt < 2; ++mt) {
#pragma unroll
      for (int r = 0; r < 8; ++r) {
        float s0v = sacc[mt][0][r] * 0.125f;
        float s1v = sacc[mt][1][r] * 0.125f;
        float bm = fmaxf(s0v, s1v);
        bm = fmaxf(bm, __shfl_xor(bm, 1, 16));
        bm = fmaxf(bm, __shfl_xor(bm, 2, 16));
        bm = fmaxf(bm, __shfl_xor(bm, 4, 16));
        bm = fmaxf(bm, __shfl_xor(bm, 8, 16));
        float mnew = fmaxf(mrow[mt][r], bm);
        float alpha = __expf(mrow[mt][r] - mnew);
        mrow[mt][r] = mnew;
        float p0 = __expf(s0v - mnew);
        float p1 = __expf(s1v - mnew);
        sacc[mt][0][r] = p0;
        sacc[mt][1][r] = p1;
        float rs = p0 + p1;
        rs += __shfl_xor(rs, 1, 16);
        rs += __shfl_xor(rs, 2, 16);
        rs += __shfl_xor(rs, 4, 16);
        rs += __shfl_xor(rs, 8, 16);
        lrow[mt][r] = lrow[mt][r] * alpha + rs;
#pragma unroll
        for (int nt = 0; nt < 4; ++nt) acc[mt][nt][r] *= alpha;
      }
    }
    // ---- relayout P (C-layout f32 -> A-layout f16) through per-wave LDS
#pragma unroll
    for (int mt = 0; mt < 2; ++mt)
#pragma unroll
      for (int nt = 0; nt < 2; ++nt)
#pragma unroll
        for (int r = 0; r < 8; ++r)
          pbase[(mt * 16 + r + 8 * hi) * 32 + nt * 16 + ln] = (_Float16)sacc[mt][nt][r];
    asm volatile("s_wait_dscnt 0" ::: "memory");
    // ---- O += P V
    v16h vf[4];
#pragma unroll
    for (int nt = 0; nt < 4; ++nt) vf[nt] = load_frag16x32(Vh, nt * 16, SLEN, key0);
    v16h pf[2];
#pragma unroll
    for (int mt = 0; mt < 2; ++mt) pf[mt] = load_frag16x32(pbase, mt * 16, 32, 0);
#pragma unroll
    for (int mt = 0; mt < 2; ++mt)
#pragma unroll
      for (int nt = 0; nt < 4; ++nt) acc[mt][nt] = wmma_f16(pf[mt], vf[nt], acc[mt][nt]);
  }

  // ---- normalize + merge heads into concat buffer
#pragma unroll
  for (int mt = 0; mt < 2; ++mt) {
#pragma unroll
    for (int r = 0; r < 8; ++r) {
      float inv = 1.0f / lrow[mt][r];
      int row = rowBase + s0 + mt * 16 + r + 8 * hi;
#pragma unroll
      for (int nt = 0; nt < 4; ++nt) {
        int col = colOff + qkCol + nt * 16 + ln;
        Y[(size_t)row * 2048 + col] = (_Float16)(acc[mt][nt][r] * inv);
      }
    }
  }
}

// ---------------------------------------------------------------------------
extern "C" void kernel_launch(void* const* d_in, const int* in_sizes, int n_in,
                              void* d_out, int out_size, void* d_ws, size_t ws_size,
                              hipStream_t stream) {
  (void)in_sizes; (void)n_in; (void)out_size; (void)ws_size;
  const float* f1 = (const float*)d_in[0];
  const float* f2 = (const float*)d_in[1];
  const float* W[6];
  const float* bv[6];
  for (int j = 0; j < 6; ++j) {           // qs, ks, vs, qc, kc, vc
    W[j] = (const float*)d_in[2 + 2 * j];
    bv[j] = (const float*)d_in[3 + 2 * j];
  }
  const float* Wo = (const float*)d_in[14];
  const float* bo = (const float*)d_in[15];
  float* out = (float*)d_out;

  // workspace carve-up (all f16)
  size_t off = 0;
  auto alloc = [&](size_t elems) {
    _Float16* p = (_Float16*)((char*)d_ws + off);
    off += (elems * sizeof(_Float16) + 255) & ~(size_t)255;
    return p;
  };
  const size_t nAct = (size_t)MROWS * 1024;   // 4096x1024
  const size_t nWp = 512 * 1024;              // projection weight
  _Float16* f1h = alloc(nAct);
  _Float16* f2h = alloc(nAct);
  _Float16* wph[6];
  for (int j = 0; j < 6; ++j) wph[j] = alloc(nWp);
  _Float16* woh = alloc((size_t)1024 * 2048);
  _Float16* qsb = alloc((size_t)MROWS * 512);
  _Float16* ksb = alloc((size_t)MROWS * 512);
  _Float16* qcb = alloc((size_t)MROWS * 512);
  _Float16* kcb = alloc((size_t)MROWS * 512);
  _Float16* yb  = alloc((size_t)MROWS * 2048);  // [merge(ys)|merge(yc)|vs|vc]
  _Float16* vtS = alloc((size_t)16 * 64 * SLEN);
  _Float16* vtC = alloc((size_t)16 * 64 * SLEN);

  // 1) f32 -> f16 conversions
  cvt_f32_f16<<<(int)(nAct / 256), 256, 0, stream>>>(f1, f1h, (int)nAct);
  cvt_f32_f16<<<(int)(nAct / 256), 256, 0, stream>>>(f2, f2h, (int)nAct);
  for (int j = 0; j < 6; ++j)
    cvt_f32_f16<<<(int)(nWp / 256), 256, 0, stream>>>(W[j], wph[j], (int)nWp);
  cvt_f32_f16<<<(1024 * 2048) / 256, 256, 0, stream>>>(Wo, woh, 1024 * 2048);

  // 2) projections: relu(x W^T + b), N=512, K=1024
  dim3 gp(MROWS / 128, 512 / 128);
  gemm_bias_relu<1><<<gp, 256, 0, stream>>>(f1h, 1024, wph[0], 1024, bv[0], qsb, 512, 1024);
  gemm_bias_relu<1><<<gp, 256, 0, stream>>>(f1h, 1024, wph[1], 1024, bv[1], ksb, 512, 1024);
  gemm_bias_relu<1><<<gp, 256, 0, stream>>>(f1h, 1024, wph[2], 1024, bv[2], yb + 1024, 2048, 1024);
  gemm_bias_relu<1><<<gp, 256, 0, stream>>>(f1h, 1024, wph[3], 1024, bv[3], qcb, 512, 1024);
  gemm_bias_relu<1><<<gp, 256, 0, stream>>>(f2h, 1024, wph[4], 1024, bv[4], kcb, 512, 1024);
  gemm_bias_relu<1><<<gp, 256, 0, stream>>>(f2h, 1024, wph[5], 1024, bv[5], yb + 1536, 2048, 1024);

  // 3) transpose V slices for PV fragments
  int nT = BDIM * SLEN * 512;
  vtrans<<<nT / 256, 256, 0, stream>>>(yb, vtS, 1024);
  vtrans<<<nT / 256, 256, 0, stream>>>(yb, vtC, 1536);

  // 4) attention (self then cross), merged heads -> yb cols 0 / 512
  attn_kernel<<<128, 256, 0, stream>>>(qsb, ksb, vtS, yb, 0);
  attn_kernel<<<128, 256, 0, stream>>>(qcb, kcb, vtC, yb, 512);

  // 5) output projection: relu(y Wo^T + bo), N=1024, K=2048, f32 out
  dim3 go(MROWS / 128, 1024 / 128);
  gemm_bias_relu<0><<<go, 256, 0, stream>>>(yb, 2048, woh, 2048, bo, out, 1024, 2048);
}
